// QLoRAMiniSAM_31628139168310
// MI455X (gfx1250) — matrix operand order repaired and checked by reading, not compiled, not run on hardware
//
#include <hip/hip_runtime.h>

// ---------------------------------------------------------------------------
// QLoRA SAM qkv projection for MI455X (gfx1250, wave32, WMMA + TDM)
//   Y[M,N] = X[M,K] @ Weff[N,K]^T,  M=32768, N=2304, K=768
//   Weff = NF4[w_idx]*scale + (alpha/r) * lora_b @ lora_a   (rank-16 fold, exact)
// GEMM: v_wmma_f32_16x16x32_bf16, 128x128x32 block tiles, double-buffered LDS.
// W tiles arrive via the Tensor Data Mover (tensor_load_to_lds, TENSORcnt);
// X tiles are converted f32->bf16 in-flight (v_cvt_pk_bf16_f32 co-executes).
// ---------------------------------------------------------------------------

typedef __bf16 bf16_t;
typedef __attribute__((ext_vector_type(16))) bf16_t       v16bf;
typedef __attribute__((ext_vector_type(8)))  bf16_t       v8bf;
typedef __attribute__((ext_vector_type(8)))  float        v8f;
typedef __attribute__((ext_vector_type(4)))  float        v4f;
typedef __attribute__((ext_vector_type(4)))  unsigned int v4u;
typedef __attribute__((ext_vector_type(8)))  int          v8i;
typedef __attribute__((ext_vector_type(4)))  int          v4i;

#define QD 768          // K (inner dim)
#define QO 2304         // N
#define QBLK 64         // NF4 block size
#define LORA_R 16
#define LORA_SCALE 2.0f // alpha/r = 32/16

#define BM 128
#define BN 128
#define BK 32
#define LDT 40          // padded LDS row stride (bf16) = 64B row + 16B pad, 16B-aligned
#define NTHREADS 256

__device__ __constant__ float c_nf4[16] = {
    -1.0f, -0.6961928009986877f, -0.5250730514526367f, -0.39491748809814453f,
    -0.28444138169288635f, -0.18477343022823334f, -0.09105003625154495f, 0.0f,
    0.07958029955625534f, 0.16093020141124725f, 0.24611230194568634f,
    0.33791524171829224f, 0.44070982933044434f, 0.5626170039176941f,
    0.7229568362236023f, 1.0f};

// ---- Kernel 1: dequant NF4 + fold LoRA, emit bf16 Weff[N,K] row-major -----
__global__ __launch_bounds__(NTHREADS) void build_weff_kernel(
    const int* __restrict__ w_idx, const float* __restrict__ scales,
    const float* __restrict__ lora_a, const float* __restrict__ lora_b,
    bf16_t* __restrict__ weff)
{
    const int i = blockIdx.x * NTHREADS + threadIdx.x;   // element over O*D
    const int o = i / QD;
    const int d = i - o * QD;
    const float w = c_nf4[w_idx[i] & 15] * scales[o * (QD / QBLK) + (d >> 6)];
    float acc = 0.0f;
#pragma unroll
    for (int r = 0; r < LORA_R; ++r)
        acc += lora_b[o * LORA_R + r] * lora_a[r * QD + d];
    weff[i] = (bf16_t)(w + LORA_SCALE * acc);
}

// ---- TDM: DMA one 128x32 bf16 W tile into padded LDS (rows stride 80B) ----
// D# layout per CDNA5 ISA.
__device__ __forceinline__ void tdm_load_w_tile(unsigned lds_off, const bf16_t* gptr)
{
    const unsigned long long ga = (unsigned long long)(size_t)gptr;
    v4u g0;
    g0[0] = 1u;                                   // count=1 valid user descriptor
    g0[1] = lds_off;                              // lds_addr (bytes)
    g0[2] = (unsigned)ga;                         // global_addr[31:0]
    g0[3] = (unsigned)((ga >> 32) & 0x01FFFFFFu)  // global_addr[56:32]
          | 0x80000000u;                          // type=2 ("image")
    v8i g1;
    g1[0] = (int)((1u << 16)                      // data_size = 2 bytes
                | (1u << 20)                      // pad_enable
                | (3u << 22)                      // pad_interval: 16 DWORDs (64B row)
                | (3u << 25));                    // pad_amount:   4 DWORDs (16B pad)
    g1[1] = (int)(((unsigned)QD & 0xFFFFu) << 16);  // tensor_dim0[15:0]=768
    g1[2] = (int)(((unsigned)QO & 0xFFFFu) << 16);  // dim0[31:16]=0 | tensor_dim1[15:0]=2304
    g1[3] = (int)((unsigned)BK << 16);              // dim1[31:16]=0 | tile_dim0=32
    g1[4] = (int)BN;                                // tile_dim1=128, tile_dim2=0
    g1[5] = (int)QD;                                // tensor_dim0_stride lo32 = 768
    g1[6] = 0;                                      // stride hi16 | dim1_stride lo16
    g1[7] = 0;
    const v4i z4 = {0, 0, 0, 0};                    // 2D tensor: groups 2/3 unused
#if __clang_major__ >= 23
    const v8i z8 = {0, 0, 0, 0, 0, 0, 0, 0};
    __builtin_amdgcn_tensor_load_to_lds(g0, g1, z4, z4, z8, 0);
#else
    __builtin_amdgcn_tensor_load_to_lds(g0, g1, z4, z4, 0);
#endif
}

// ---- Kernel 2: tiled WMMA GEMM  Y = X @ Weff^T ----------------------------
__global__ __launch_bounds__(NTHREADS) void qlora_gemm_kernel(
    const float* __restrict__ X, const bf16_t* __restrict__ W,
    float* __restrict__ Y, int M, int N, int K)
{
    __shared__ bf16_t sA[2][BM * LDT];   // X tile, [row][k], bf16 (converted in-flight)
    __shared__ bf16_t sB[2][BN * LDT];   // W tile, [n][k],  bf16 (filled by TDM)

    const int tid  = threadIdx.x;
    const int lane = tid & 31;
    const int wid  = tid >> 5;           // 8 waves
    const int wm   = wid >> 2;           // 0..1 -> 64-row band
    const int wn   = wid & 3;            // 0..3 -> 32-col band

    const int bm = blockIdx.y * BM;
    const int bn = blockIdx.x * BN;      // x fastest => N-blocks adjacent, X panel reuse in L2

    // LDS byte offsets (generic LDS address low 32 bits == LDS offset)
    const unsigned sb_off0 = (unsigned)(size_t)&sB[0][0];
    const unsigned sb_off1 = (unsigned)(size_t)&sB[1][0];

    // X staging coords: each thread owns 16 contiguous k-elems of one tile row
    const int lrow  = tid >> 1;          // 0..127
    const int lhalf = (tid & 1) << 4;    // 0 or 16
    const float* xg = X + (size_t)(bm + lrow) * K + lhalf;
    const bf16_t* wblk = W + (size_t)bn * K;

    v8f acc[4][2];
#pragma unroll
    for (int mi = 0; mi < 4; ++mi)
#pragma unroll
        for (int ni = 0; ni < 2; ++ni)
#pragma unroll
            for (int e = 0; e < 8; ++e) acc[mi][ni][e] = 0.0f;

    // fragment lane addressing (contiguous k = 16*(lane/16) + s, same order in A and B)
    const int ar = lane & 15;              // row (A) / col (B) within 16
    const int ak = (lane >> 4) << 4;       // A: k-base 0 or 16
    const int bk = (lane >> 4) << 4;       // B: k-base 0 or 16

    union Frag { v16bf v; v8bf h[2]; };

    v4f xr[4];
    const int NT = K / BK;                 // 24

    // --- prologue: TDM W tile 0, stage X tile 0 ---
    if (wid == 0) tdm_load_w_tile(sb_off0, wblk);
    {
        const v4f* xp = (const v4f*)xg;
#pragma unroll
        for (int j = 0; j < 4; ++j) xr[j] = xp[j];
        v8bf t0, t1;
#pragma unroll
        for (int j = 0; j < 4; ++j) {
            t0[j]     = (bf16_t)xr[0][j];
            t0[j + 4] = (bf16_t)xr[1][j];
            t1[j]     = (bf16_t)xr[2][j];
            t1[j + 4] = (bf16_t)xr[3][j];
        }
        bf16_t* pa = &sA[0][lrow * LDT + lhalf];
        *(v8bf*)pa = t0; *(v8bf*)(pa + 8) = t1;
    }
    if (wid == 0) __builtin_amdgcn_s_wait_tensorcnt(0);
    __syncthreads();

    int buf = 0;
    for (int kt = 0; kt < NT; ++kt) {
        const int k0n = (kt + 1) * BK;
        const bool more = (k0n < K);

        // issue next-tile transfers early (latency hidden by WMMA below)
        if (more) {
            const v4f* xp = (const v4f*)(xg + k0n);
#pragma unroll
            for (int j = 0; j < 4; ++j) xr[j] = xp[j];
            if (wid == 0)
                tdm_load_w_tile(buf ? sb_off0 : sb_off1, wblk + k0n);
        }

        // --- compute on current buffer ---
        Frag afr[4], bfr[2];
#pragma unroll
        for (int mi = 0; mi < 4; ++mi) {
            const bf16_t* p = &sA[buf][(wm * 64 + mi * 16 + ar) * LDT + ak];
            afr[mi].h[0] = *(const v8bf*)p;          // K = ak .. ak+7
            afr[mi].h[1] = *(const v8bf*)(p + 8);    // K = ak+8 .. ak+15
        }
#pragma unroll
        for (int ni = 0; ni < 2; ++ni) {
            const bf16_t* p = &sB[buf][(wn * 32 + ni * 16 + ar) * LDT + bk];
            bfr[ni].h[0] = *(const v8bf*)p;          // K = bk .. bk+7
            bfr[ni].h[1] = *(const v8bf*)(p + 8);    // K = bk+8 .. bk+15
        }
#pragma unroll
        for (int mi = 0; mi < 4; ++mi)
#pragma unroll
            for (int ni = 0; ni < 2; ++ni)
                acc[mi][ni] = __builtin_amdgcn_wmma_f32_16x16x32_bf16(
                    false, afr[mi].v, false, bfr[ni].v,
                    (short)0, acc[mi][ni], false, false);

        // --- stage next X tile; TDM completion folds into the existing barrier ---
        if (more) {
            v8bf t0, t1;
#pragma unroll
            for (int j = 0; j < 4; ++j) {
                t0[j]     = (bf16_t)xr[0][j];
                t0[j + 4] = (bf16_t)xr[1][j];
                t1[j]     = (bf16_t)xr[2][j];
                t1[j + 4] = (bf16_t)xr[3][j];
            }
            bf16_t* pa = &sA[buf ^ 1][lrow * LDT + lhalf];
            *(v8bf*)pa = t0; *(v8bf*)(pa + 8) = t1;
            if (wid == 0) __builtin_amdgcn_s_wait_tensorcnt(0);
            __syncthreads();
            buf ^= 1;
        }
    }

    // --- writeback: C/D layout = VGPR e: rows e (lanes 0-15) / e+8 (16-31) ---
    // Nontemporal: Y is a 302MB write-once stream; keep L2 for X panel reuse.
#pragma unroll
    for (int mi = 0; mi < 4; ++mi) {
#pragma unroll
        for (int ni = 0; ni < 2; ++ni) {
            const int row0 = bm + wm * 64 + mi * 16 + ((lane >> 4) << 3);
            const int col  = bn + wn * 32 + ni * 16 + (lane & 15);
            float* yp = Y + (size_t)row0 * N + col;
#pragma unroll
            for (int e = 0; e < 8; ++e)
                __builtin_nontemporal_store(acc[mi][ni][e], yp + (size_t)e * N);
        }
    }
}

// ---------------------------------------------------------------------------
extern "C" void kernel_launch(void* const* d_in, const int* in_sizes, int n_in,
                              void* d_out, int out_size, void* d_ws, size_t ws_size,
                              hipStream_t stream)
{
    const float* x      = (const float*)d_in[0];   // [8,4096,768]
    const int*   w_idx  = (const int*)  d_in[1];   // [2304,768]
    const float* scales = (const float*)d_in[2];   // [2304,12]
    const float* lora_a = (const float*)d_in[3];   // [16,768]
    const float* lora_b = (const float*)d_in[4];   // [2304,16]
    float*       out    = (float*)d_out;           // [8,4096,2304]

    bf16_t* weff = (bf16_t*)d_ws;                  // 2304*768*2 = 3.4 MB scratch

    const int M = in_sizes[0] / QD;                // 32768
    const int N = QO;                              // 2304
    const int K = QD;                              // 768

    // 1) dequant + fold LoRA into bf16 Weff
    const int OD = QO * QD;                        // 1,769,472 (divisible by 256)
    build_weff_kernel<<<OD / NTHREADS, NTHREADS, 0, stream>>>(
        w_idx, scales, lora_a, lora_b, weff);

    // 2) WMMA GEMM; grid.x = N-blocks (fast dim -> L2 reuse of X panels)
    dim3 grid(N / BN, M / BM);                     // (18, 256)
    qlora_gemm_kernel<<<grid, NTHREADS, 0, stream>>>(x, weff, out, M, N, K);
}
